// EGNNStab_2886218023088
// MI455X (gfx1250) — matrix operand order repaired and checked by reading, not compiled, not run on hardware
//
#include <hip/hip_runtime.h>
#include <hip/hip_bf16.h>

// ---------------------------------------------------------------------------
// EGNN center-attention + gated MLP, restructured for MI455X (gfx1250).
//  - attention algebraically collapsed to two memory-bound passes over h
//  - gated MLP = 3 fused DxD GEMMs in bf16 via v_wmma_f32_16x16x32_bf16
//  - weight tiles staged LDS-direct via global_load_async_to_lds_b128
//    (ASYNCcnt), double-buffered so the DMA hides behind the WMMA stream
// ---------------------------------------------------------------------------

#define NN    65536
#define DD    1024
#define HH_   16
#define HD_   64
#define KK    4096
#define MT    64          // rows per block in MLP kernel
#define EPSL  1e-5f

typedef __attribute__((ext_vector_type(16))) __bf16 v16bf;
typedef __attribute__((ext_vector_type(8)))  float  v8f;

union ABfrag { v16bf v; uint4 q[2]; };

__device__ __forceinline__ unsigned f2bf(float f) {
    unsigned u = __float_as_uint(f);
    unsigned r = (u + 0x7fffu + ((u >> 16) & 1u)) >> 16;   // RNE
    return r & 0xffffu;
}
__device__ __forceinline__ float bf2f(unsigned short s) {
    return __uint_as_float(((unsigned)s) << 16);
}

// async DMA: 16B per lane, global -> LDS, tracked with ASYNCcnt.
// LDS operand is the byte offset = low 32 bits of the flat pointer
// (flat-LDS aperture truncates to the LDS address, ISA 10.2).
__device__ __forceinline__ void async_b128(void* lds, const void* gmem) {
    unsigned ldsoff = (unsigned)(size_t)lds;
    asm volatile("global_load_async_to_lds_b128 %0, %1, off"
                 :: "v"(ldsoff), "v"(gmem) : "memory");
}
__device__ __forceinline__ void wait_async0() {
    asm volatile("s_wait_asynccnt 0" ::: "memory");
}

// ------------------------------ tiny helpers -------------------------------
__global__ void k_zero(float* p, int n) {
    int i = blockIdx.x * blockDim.x + threadIdx.x;
    if (i < n) p[i] = 0.0f;
}

// scatter local_bias rows into full_bias (N,20)
__global__ void k_scatter(const float* __restrict__ lb, const int* __restrict__ idx,
                          float* __restrict__ fb) {
    int i = blockIdx.x * blockDim.x + threadIdx.x;
    if (i >= KK * 20) return;
    int j = i / 20, f = i % 20;
    fb[(size_t)idx[j] * 20 + f] = lb[i];
}

// convert left halves of gate_W / fuse_W1 and all of fuse_W2 to bf16
__global__ void k_cvtW(const float* __restrict__ gw, const float* __restrict__ fw1,
                       const float* __restrict__ fw2,
                       unsigned short* __restrict__ Gb, unsigned short* __restrict__ F1b,
                       unsigned short* __restrict__ W2b) {
    int i = blockIdx.x * blockDim.x + threadIdx.x;
    if (i >= 3 * DD * DD) return;
    int which = i >> 20;            // DD*DD == 1<<20
    int r = i & ((1 << 20) - 1);
    int j = r >> 10, k = r & 1023;
    if (which == 0)      Gb[r]  = (unsigned short)f2bf(gw[(size_t)j * 2048 + k]);
    else if (which == 1) F1b[r] = (unsigned short)f2bf(fw1[(size_t)j * 2048 + k]);
    else                 W2b[r] = (unsigned short)f2bf(fw2[r]);
}

// ------------------- center row: LN, q, folded qk vectors ------------------
__global__ __launch_bounds__(1024) void k_center(
        const float* __restrict__ h, const int* __restrict__ cptr,
        const float* __restrict__ ncw, const float* __restrict__ ncb,
        const float* __restrict__ naw, const float* __restrict__ nab,
        const float* __restrict__ Wq, const float* __restrict__ Wk,
        float* __restrict__ wqk, float* __restrict__ swqk, float* __restrict__ bqk) {
    __shared__ float red[1024], red2[1024], hcn[1024], qs[1024];
    int tid = threadIdx.x;
    int c = *cptr;
    float x = h[(size_t)c * DD + tid];
    red[tid] = x; red2[tid] = x * x; __syncthreads();
    for (int s = 512; s > 0; s >>= 1) {
        if (tid < s) { red[tid] += red[tid + s]; red2[tid] += red2[tid + s]; }
        __syncthreads();
    }
    float mu = red[0] * (1.0f / DD);
    float var = red2[0] * (1.0f / DD) - mu * mu;
    float rs = rsqrtf(var + EPSL);
    hcn[tid] = (x - mu) * rs * ncw[tid] + ncb[tid];
    __syncthreads();
    float q = 0.0f;
    const float* wr = Wq + (size_t)tid * DD;
    for (int d = 0; d < DD; ++d) q += wr[d] * hcn[d];
    qs[tid] = q;
    __syncthreads();
    // qk[h,d] = sum_e q[h*64+e]*Wk[h*64+e][d]; fold LN-all affine and 1/8 scale
    for (int hh = 0; hh < HH_; ++hh) {
        float qk = 0.0f;
        for (int e = 0; e < HD_; ++e)
            qk += qs[hh * HD_ + e] * Wk[(size_t)(hh * HD_ + e) * DD + tid];
        float wv = qk * naw[tid] * 0.125f;
        wqk[hh * DD + tid] = wv;
        atomicAdd(&swqk[hh], wv);
        atomicAdd(&bqk[hh], qk * nab[tid] * 0.125f);
    }
}

// ------------------ pass A: per-node LN stats + 16 logits ------------------
__global__ __launch_bounds__(256) void k_passA(
        const float* __restrict__ h, const float* __restrict__ rbf,
        const float* __restrict__ ssp, const float* __restrict__ fb,
        const float* __restrict__ Wbias, const float* __restrict__ wqkG,
        const float* __restrict__ swqk, const float* __restrict__ bqk,
        float* __restrict__ logits, float* __restrict__ muA, float* __restrict__ rsA) {
    extern __shared__ float sm[];
    float* wq = sm;                  // 16*1024
    float* dred = sm + HH_ * DD;     // 8*16
    int tid = threadIdx.x, lane = tid & 31, wave = tid >> 5;
    for (int i = tid; i < HH_ * DD; i += 256) wq[i] = wqkG[i];
    __syncthreads();
    int n = blockIdx.x * 8 + wave;
    const float* hp = h + (size_t)n * DD;
    float sum = 0.0f, sq = 0.0f, dot[HH_];
#pragma unroll
    for (int hh = 0; hh < HH_; ++hh) dot[hh] = 0.0f;
    for (int ch = 0; ch < 32; ++ch) {
        int d = ch * 32 + lane;
        float x = hp[d];
        sum += x; sq += x * x;
#pragma unroll
        for (int hh = 0; hh < HH_; ++hh) dot[hh] += x * wq[hh * DD + d];
    }
#pragma unroll
    for (int m = 16; m >= 1; m >>= 1) {
        sum += __shfl_xor(sum, m, 32);
        sq  += __shfl_xor(sq,  m, 32);
#pragma unroll
        for (int hh = 0; hh < HH_; ++hh) dot[hh] += __shfl_xor(dot[hh], m, 32);
    }
    float mu = sum * (1.0f / DD);
    float var = sq * (1.0f / DD) - mu * mu;
    float rs = rsqrtf(var + EPSL);
    if (lane == 0) {
        muA[n] = mu; rsA[n] = rs;
#pragma unroll
        for (int hh = 0; hh < HH_; ++hh) dred[wave * HH_ + hh] = dot[hh];
    }
    __syncthreads();
    if (lane < HH_) {
        int hh = lane;
        float dv = dred[wave * HH_ + hh];
        const float* wb = Wbias + hh * 40;
        float bias = 0.0f;
#pragma unroll
        for (int i = 0; i < 16; ++i) bias += wb[i] * rbf[(size_t)n * 16 + i];
#pragma unroll
        for (int i = 0; i < 4; ++i)  bias += wb[16 + i] * ssp[(size_t)n * 4 + i];
#pragma unroll
        for (int i = 0; i < 20; ++i) bias += wb[20 + i] * fb[(size_t)n * 20 + i];
        logits[(size_t)hh * NN + n] = rs * (dv - mu * swqk[hh]) + bqk[hh] + bias;
    }
}

// --------------- softmax per head; fold rs into coeffs; mus ----------------
__global__ __launch_bounds__(1024) void k_softmax(
        float* __restrict__ logits, const float* __restrict__ muA,
        const float* __restrict__ rsA, float* __restrict__ musArr) {
    __shared__ float red[1024];
    int hh = blockIdx.x, tid = threadIdx.x;
    float* lp = logits + (size_t)hh * NN;
    float mx = -1e30f;
    for (int n = tid; n < NN; n += 1024) mx = fmaxf(mx, lp[n]);
    red[tid] = mx; __syncthreads();
    for (int s = 512; s > 0; s >>= 1) {
        if (tid < s) red[tid] = fmaxf(red[tid], red[tid + s]);
        __syncthreads();
    }
    float M = red[0]; __syncthreads();
    float se = 0.0f;
    for (int n = tid; n < NN; n += 1024) se += __expf(lp[n] - M);
    red[tid] = se; __syncthreads();
    for (int s = 512; s > 0; s >>= 1) {
        if (tid < s) red[tid] += red[tid + s];
        __syncthreads();
    }
    float inv = 1.0f / red[0]; __syncthreads();
    float cm = 0.0f;
    for (int n = tid; n < NN; n += 1024) {
        float a = __expf(lp[n] - M) * inv;
        float cc = a * rsA[n];      // fold rs_n
        lp[n] = cc;
        cm += cc * muA[n];
    }
    red[tid] = cm; __syncthreads();
    for (int s = 512; s > 0; s >>= 1) {
        if (tid < s) red[tid] += red[tid + s];
        __syncthreads();
    }
    if (tid == 0) musArr[hh] = red[0];
}

// ------------- pass B: S[h,d] = sum_n coeff[h,n]*h[n,d] (atomics) ----------
__global__ __launch_bounds__(256) void k_passB(
        const float* __restrict__ h, const float* __restrict__ coeff,
        float* __restrict__ Svec) {
    int tid = threadIdx.x;
    int base = blockIdx.x * 128;
    float acc[HH_][4] = {};
    for (int rr = 0; rr < 128; ++rr) {
        int n = base + rr;
        float cv[HH_];
#pragma unroll
        for (int hh = 0; hh < HH_; ++hh) cv[hh] = coeff[(size_t)hh * NN + n];
        const float* hp = h + (size_t)n * DD;
#pragma unroll
        for (int qd = 0; qd < 4; ++qd) {
            float x = hp[tid + qd * 256];
#pragma unroll
            for (int hh = 0; hh < HH_; ++hh) acc[hh][qd] += cv[hh] * x;
        }
    }
#pragma unroll
    for (int hh = 0; hh < HH_; ++hh)
#pragma unroll
        for (int qd = 0; qd < 4; ++qd)
            atomicAdd(&Svec[hh * DD + tid + qd * 256], acc[hh][qd]);
}

// ---- post: out_center, Wo, h_c_new, MLP constant vectors cg / cf ----------
__global__ __launch_bounds__(1024) void k_post(
        const float* __restrict__ h, const int* __restrict__ cptr,
        const float* __restrict__ naw, const float* __restrict__ nab,
        const float* __restrict__ Wv, const float* __restrict__ Wo,
        const float* __restrict__ gateW, const float* __restrict__ gateB,
        const float* __restrict__ fuseW1, const float* __restrict__ fuseB1,
        const float* __restrict__ Svec, const float* __restrict__ musArr,
        float* __restrict__ hcnew, float* __restrict__ cgv, float* __restrict__ cfv) {
    extern __shared__ float sm[];
    float* s    = sm;                  // 16*1024
    float* outc = sm + HH_ * DD;       // 1024
    float* hcn  = outc + DD;           // 1024
    int tid = threadIdx.x;
    int c = *cptr;
#pragma unroll
    for (int hh = 0; hh < HH_; ++hh)
        s[hh * DD + tid] = naw[tid] * (Svec[hh * DD + tid] - musArr[hh]) + nab[tid];
    __syncthreads();
    int hh = tid >> 6;
    float oc = 0.0f;
    const float* wr = Wv + (size_t)tid * DD;
    const float* sp = s + hh * DD;
    for (int d = 0; d < DD; ++d) oc += wr[d] * sp[d];
    outc[tid] = oc;
    __syncthreads();
    float del = 0.0f;
    const float* wo = Wo + (size_t)tid * DD;
    for (int i = 0; i < DD; ++i) del += wo[i] * outc[i];
    float hv = h[(size_t)c * DD + tid] + 0.5f * del;
    hcnew[tid] = hv; hcn[tid] = hv;
    __syncthreads();
    float a = gateB[tid], b = fuseB1[tid];
    const float* gw = gateW + (size_t)tid * 2048 + DD;   // right half of gate_W
    const float* fw = fuseW1 + (size_t)tid * 2048 + DD;  // right half of fuse_W1
    for (int i = 0; i < DD; ++i) { a += gw[i] * hcn[i]; b += fw[i] * hcn[i]; }
    cgv[tid] = a; cfv[tid] = b;
}

// -------------------- the big one: fused gated MLP, bf16 WMMA --------------
// per block: 64 rows; three sweeps (gate, fuse1, fuse2) of 4 x 256-col tiles;
// 8 waves = 4 row-groups x 2 col-halves, each wave 16x128 = 8 WMMA C tiles.
// B tiles DMA'd into LDS (double-buffered) so the copy hides behind WMMA.
__global__ __launch_bounds__(256) void k_mlp(
        const float* __restrict__ h, const float* __restrict__ hcnew,
        const float* __restrict__ cg, const float* __restrict__ cf,
        const float* __restrict__ b2,
        const unsigned short* __restrict__ Gb, const unsigned short* __restrict__ F1b,
        const unsigned short* __restrict__ W2b, const int* __restrict__ cptr,
        float* __restrict__ out) {
    extern __shared__ char smem[];
    unsigned short* Ut = (unsigned short*)smem;        // [MT][1024] silu(fuse1) bf16
    unsigned short* Gt = Ut + MT * DD;                 // [MT][1024] sigmoid gate bf16
    unsigned short* As = Gt + MT * DD;                 // [2][MT][32]  A stage (dbl buf)
    unsigned short* Bs = As + 2 * MT * 32;             // [2][256][32] B stage (dbl buf)

    const int tid = threadIdx.x;
    const int lane = tid & 31, wave = tid >> 5;
    const int rg = wave >> 1, cw = wave & 1;
    const int lrow = lane & 15, lhi = lane >> 4;
    const int rowbase = blockIdx.x * MT;
    const int c = *cptr;

    for (int sweep = 0; sweep < 3; ++sweep) {
        const unsigned short* W = (sweep == 0) ? Gb : (sweep == 1) ? F1b : W2b;
        for (int cb = 0; cb < 4; ++cb) {
            v8f acc[8] = {};

            // ---- staging helpers (uniform across block) ----
            auto stageB = [&](int ks, int buf) {
                // 256 cols x 32 k of bf16 weight, DMA straight into LDS
                const unsigned short* src = W + (size_t)(cb * 256 + tid) * DD + ks * 32;
                unsigned short* dst = Bs + buf * 256 * 32 + tid * 32;
                async_b128(dst,      src);
                async_b128(dst + 8,  src + 8);
                async_b128(dst + 16, src + 16);
                async_b128(dst + 24, src + 24);
            };
            auto stageA = [&](int ks, int buf) {
                // 64 rows x 32 k of h, f32 -> bf16 (needs VALU convert)
                int row = tid >> 2, seg = tid & 3;
                const float* src = h + (size_t)(rowbase + row) * DD + ks * 32 + seg * 8;
                float4 f0 = ((const float4*)src)[0];
                float4 f1 = ((const float4*)src)[1];
                uint4 o;
                o.x = f2bf(f0.x) | (f2bf(f0.y) << 16);
                o.y = f2bf(f0.z) | (f2bf(f0.w) << 16);
                o.z = f2bf(f1.x) | (f2bf(f1.y) << 16);
                o.w = f2bf(f1.z) | (f2bf(f1.w) << 16);
                *(uint4*)&As[buf * MT * 32 + row * 32 + seg * 8] = o;
            };

            // prologue: fill buffer 0
            stageB(0, 0);
            if (sweep < 2) stageA(0, 0);

            for (int ks = 0; ks < 32; ++ks) {
                const int buf = ks & 1;
                wait_async0();          // this wave's DMA into buf has landed
                __syncthreads();        // everyone's stores/DMA visible
                if (ks + 1 < 32) {      // overlap next-step staging with WMMA
                    stageB(ks + 1, buf ^ 1);
                    if (sweep < 2) stageA(ks + 1, buf ^ 1);
                }
                // A fragment (ISA 16-bit A 16x32 layout)
                ABfrag a;
                if (sweep == 2) {
                    const unsigned short* ab = Ut + (size_t)(rg * 16 + lrow) * DD + ks * 32;
                    a.q[0] = *(const uint4*)(ab + lhi * 8);
                    a.q[1] = *(const uint4*)(ab + 16 + lhi * 8);
                } else {
                    const unsigned short* ab = As + buf * MT * 32 + (rg * 16 + lrow) * 32;
                    a.q[0] = *(const uint4*)(ab + lhi * 8);
                    a.q[1] = *(const uint4*)(ab + 16 + lhi * 8);
                }
#pragma unroll
                for (int t = 0; t < 8; ++t) {
                    ABfrag b;
                    const unsigned short* bb =
                        Bs + buf * 256 * 32 + (cw * 128 + t * 16 + lrow) * 32;
                    b.q[0] = *(const uint4*)(bb + lhi * 8);
                    b.q[1] = *(const uint4*)(bb + 16 + lhi * 8);
                    acc[t] = __builtin_amdgcn_wmma_f32_16x16x32_bf16(
                        false, a.v, false, b.v, (short)0, acc[t], false, false);
                }
            }
            __syncthreads();
            // epilogue for this 64x256 tile
#pragma unroll
            for (int t = 0; t < 8; ++t) {
                int j = cb * 256 + cw * 128 + t * 16 + lrow;
#pragma unroll
                for (int r = 0; r < 8; ++r) {
                    int row = rg * 16 + lhi * 8 + r;
                    float x = acc[t][r];
                    if (sweep == 0) {
                        float g = 1.0f / (1.0f + __expf(-(x + cg[j])));
                        Gt[row * DD + j] = (unsigned short)f2bf(g);
                    } else if (sweep == 1) {
                        float z = x + cf[j];
                        float u = z / (1.0f + __expf(-z));
                        Ut[row * DD + j] = (unsigned short)f2bf(u);
                    } else {
                        float o = x + b2[j];
                        float g = bf2f(Gt[row * DD + j]);
                        int grow = rowbase + row;
                        float res = h[(size_t)grow * DD + j] + 0.5f * g * o;
                        out[(size_t)grow * DD + j] = (grow == c) ? hcnew[j] : res;
                    }
                }
            }
            __syncthreads();
        }
    }
}

// ---------------------------------------------------------------------------
extern "C" void kernel_launch(void* const* d_in, const int* in_sizes, int n_in,
                              void* d_out, int out_size, void* d_ws, size_t ws_size,
                              hipStream_t stream) {
    const float* h        = (const float*)d_in[0];
    const float* rbf      = (const float*)d_in[1];
    const float* ssp      = (const float*)d_in[2];
    const float* lb       = (const float*)d_in[3];
    const float* ncw      = (const float*)d_in[4];
    const float* ncb      = (const float*)d_in[5];
    const float* naw      = (const float*)d_in[6];
    const float* nab      = (const float*)d_in[7];
    const float* Wq       = (const float*)d_in[8];
    const float* Wk       = (const float*)d_in[9];
    const float* Wv       = (const float*)d_in[10];
    const float* Wo       = (const float*)d_in[11];
    const float* Wbias    = (const float*)d_in[12];
    const float* gateW    = (const float*)d_in[13];
    const float* gateB    = (const float*)d_in[14];
    const float* fuseW1   = (const float*)d_in[15];
    const float* fuseB1   = (const float*)d_in[16];
    const float* fuseW2   = (const float*)d_in[17];
    const float* fuseB2   = (const float*)d_in[18];
    const int*   nbr      = (const int*)d_in[19];
    const int*   cptr     = (const int*)d_in[20];
    float* out = (float*)d_out;

    auto au = [](size_t x) { return (x + 255) & ~(size_t)255; };
    char* p = (char*)d_ws;
    float* fb     = (float*)p; p += au((size_t)NN * 20 * 4);
    float* logits = (float*)p; p += au((size_t)HH_ * NN * 4);
    float* muA    = (float*)p; p += au((size_t)NN * 4);
    float* rsA    = (float*)p; p += au((size_t)NN * 4);
    float* wqk    = (float*)p; p += au((size_t)HH_ * DD * 4);
    float* swqk   = (float*)p; p += au(HH_ * 4);
    float* bqk    = (float*)p; p += au(HH_ * 4);
    float* musArr = (float*)p; p += au(HH_ * 4);
    float* Svec   = (float*)p; p += au((size_t)HH_ * DD * 4);
    float* hcnew  = (float*)p; p += au(DD * 4);
    float* cgv    = (float*)p; p += au(DD * 4);
    float* cfv    = (float*)p; p += au(DD * 4);
    unsigned short* Gb  = (unsigned short*)p; p += au((size_t)DD * DD * 2);
    unsigned short* F1b = (unsigned short*)p; p += au((size_t)DD * DD * 2);
    unsigned short* W2b = (unsigned short*)p; p += au((size_t)DD * DD * 2);

    // prep: zero accumulation targets, convert weights, scatter bias
    k_zero<<<(NN * 20 + 255) / 256, 256, 0, stream>>>(fb, NN * 20);
    k_zero<<<1, 256, 0, stream>>>(swqk, HH_);
    k_zero<<<1, 256, 0, stream>>>(bqk, HH_);
    k_zero<<<1, 256, 0, stream>>>(musArr, HH_);
    k_zero<<<(HH_ * DD + 255) / 256, 256, 0, stream>>>(Svec, HH_ * DD);
    k_cvtW<<<(3 * DD * DD) / 256, 256, 0, stream>>>(gateW, fuseW1, fuseW2, Gb, F1b, W2b);
    k_scatter<<<(KK * 20 + 255) / 256, 256, 0, stream>>>(lb, nbr, fb);

    // attention (collapsed)
    k_center<<<1, 1024, 0, stream>>>(h, cptr, ncw, ncb, naw, nab, Wq, Wk, wqk, swqk, bqk);
    k_passA<<<NN / 8, 256, (HH_ * DD + 8 * HH_) * sizeof(float), stream>>>(
        h, rbf, ssp, fb, Wbias, wqk, swqk, bqk, logits, muA, rsA);
    k_softmax<<<HH_, 1024, 0, stream>>>(logits, muA, rsA, musArr);
    k_passB<<<NN / 128, 256, 0, stream>>>(h, logits, Svec);
    k_post<<<1, 1024, (HH_ * DD + 2 * DD) * sizeof(float), stream>>>(
        h, cptr, naw, nab, Wv, Wo, gateW, gateB, fuseW1, fuseB1,
        Svec, musArr, hcnew, cgv, cfv);

    // fused gated MLP (bf16 WMMA), writes all of d_out (row c patched inline)
    size_t mlp_lds = (size_t)(2 * MT * DD + 2 * MT * 32 + 2 * 256 * 32) * sizeof(unsigned short);
    k_mlp<<<NN / MT, 256, mlp_lds, stream>>>(
        h, hcnew, cgv, cfv, fuseB2, Gb, F1b, W2b, cptr, out);
}
